// PerspectiveNet768x2_8821862826665
// MI455X (gfx1250) — matrix operand order repaired and checked by reading, not compiled, not run on hardware
//
#include <hip/hip_runtime.h>
#include <hip/hip_fp16.h>

typedef __attribute__((ext_vector_type(16))) _Float16 v16h;
typedef __attribute__((ext_vector_type(8)))  _Float16 v8h;
typedef __attribute__((ext_vector_type(8)))  float    v8f;

#define B_POS 16384
#define F_DIM 3072
#define H_DIM 1024
#define O_DIM 8

// GEMM tiling: block = 64(M) x 128(N), K-step 32, 8 waves of 32x32 each.
#define BM 64
#define BN 128
#define BK 32
#define NSTEPS (F_DIM / BK)   // 96

// ---------------------------------------------------------------------------
// Feature-transformer GEMM: acc[p][b,h] = feat[p][b,:] @ W[p][h,:] + bias[p][h]
// f32 inputs converted to f16 during LDS staging; f32 WMMA accumulate;
// pre-activation result stored as f16 in workspace.
// ---------------------------------------------------------------------------
__global__ __launch_bounds__(256) void ft_gemm_f16(
    const float* __restrict__ featW, const float* __restrict__ featB,
    const float* __restrict__ Ww,    const float* __restrict__ Wb,
    const float* __restrict__ bw,    const float* __restrict__ bb,
    _Float16* __restrict__ accW,     _Float16* __restrict__ accB)
{
    __shared__ __align__(32) _Float16 As[2][BM * BK];   // 2 x 4 KB
    __shared__ __align__(32) _Float16 Bs[2][BN * BK];   // 2 x 8 KB

    const int bx      = blockIdx.x;
    const int n_chunk = bx & 7;          // H/BN = 8  (fastest -> L2 reuse of feature rows)
    const int persp   = (bx >> 3) & 1;   // 0 = white net, 1 = black net
    const int m_chunk = bx >> 4;         // B/BM = 256

    const float* feat   = persp ? featB : featW;
    const float* Wmat   = persp ? Wb    : Ww;
    const float* bias   = persp ? bb    : bw;
    _Float16*    accOut = persp ? accB  : accW;

    const int m0 = m_chunk * BM;
    const int n0 = n_chunk * BN;

    const int t    = threadIdx.x;
    const int lane = t & 31;
    const int wave = t >> 5;     // 0..7
    const int wm   = wave & 1;   // M half of block tile (32 rows)
    const int wn   = wave >> 1;  // N quarter of block tile (32 cols)
    const int col  = lane & 15;  // N / M-row index inside 16x16 tile
    const int kh   = lane >> 4;  // K-half selector (CDNA5 wave32 WMMA layout)

    // Staging coordinates: A tile 64x32 f32 (8 f32/thread), B tile 128x32 f32 (16 f32/thread)
    const int ar = t >> 2,  ac = (t & 3) * 8;
    const int br = t >> 1,  bc = (t & 1) * 16;
    const float* a_src = feat + (size_t)(m0 + ar) * F_DIM + ac;
    const float* b_src = Wmat + (size_t)(n0 + br) * F_DIM + bc;

    // Accumulators, initialized with the bias of their output column.
    v8f acc[2][2];
    #pragma unroll
    for (int j = 0; j < 2; ++j) {
        const float bv = bias[n0 + wn * 32 + j * 16 + col];
        #pragma unroll
        for (int i = 0; i < 2; ++i)
            #pragma unroll
            for (int e = 0; e < 8; ++e)
                acc[i][j][e] = bv;
    }

    auto stage = [&](int buf, int k0) {
        // A: features f32 -> f16 into LDS [row][k]
        float4 fa0 = *(const float4*)(a_src + k0);
        float4 fa1 = *(const float4*)(a_src + k0 + 4);
        v8h ha;
        ha[0] = (_Float16)fa0.x; ha[1] = (_Float16)fa0.y;
        ha[2] = (_Float16)fa0.z; ha[3] = (_Float16)fa0.w;
        ha[4] = (_Float16)fa1.x; ha[5] = (_Float16)fa1.y;
        ha[6] = (_Float16)fa1.z; ha[7] = (_Float16)fa1.w;
        *(v8h*)&As[buf][ar * BK + ac] = ha;

        // B: weights f32 -> f16 into LDS [h][k] (rows are K-contiguous already)
        float4 fb0 = *(const float4*)(b_src + k0);
        float4 fb1 = *(const float4*)(b_src + k0 + 4);
        float4 fb2 = *(const float4*)(b_src + k0 + 8);
        float4 fb3 = *(const float4*)(b_src + k0 + 12);
        v8h hb0, hb1;
        hb0[0] = (_Float16)fb0.x; hb0[1] = (_Float16)fb0.y;
        hb0[2] = (_Float16)fb0.z; hb0[3] = (_Float16)fb0.w;
        hb0[4] = (_Float16)fb1.x; hb0[5] = (_Float16)fb1.y;
        hb0[6] = (_Float16)fb1.z; hb0[7] = (_Float16)fb1.w;
        hb1[0] = (_Float16)fb2.x; hb1[1] = (_Float16)fb2.y;
        hb1[2] = (_Float16)fb2.z; hb1[3] = (_Float16)fb2.w;
        hb1[4] = (_Float16)fb3.x; hb1[5] = (_Float16)fb3.y;
        hb1[6] = (_Float16)fb3.z; hb1[7] = (_Float16)fb3.w;
        *(v8h*)&Bs[buf][br * BK + bc]     = hb0;
        *(v8h*)&Bs[buf][br * BK + bc + 8] = hb1;
    };

    auto compute = [&](int buf) {
        const _Float16* A_ = As[buf];
        const _Float16* B_ = Bs[buf];
        v16h a[2], b[2];
        // A-frag (16x32 f16): lane = M row (col), kh selects K chunks {8kh..8kh+7, 16+8kh..}
        #pragma unroll
        for (int i = 0; i < 2; ++i) {
            const _Float16* row = A_ + (wm * 32 + i * 16 + col) * BK;
            v8h lo = *(const v8h*)(row + kh * 8);
            v8h hi = *(const v8h*)(row + 16 + kh * 8);
            a[i] = __builtin_shufflevector(lo, hi, 0,1,2,3,4,5,6,7,8,9,10,11,12,13,14,15);
        }
        // B-frag (32x16 f16): lane = N col (col), kh selects K chunks {8kh..8kh+7, 16+8kh..}
        #pragma unroll
        for (int j = 0; j < 2; ++j) {
            const _Float16* row = B_ + (wn * 32 + j * 16 + col) * BK;
            v8h lo = *(const v8h*)(row + kh * 8);
            v8h hi = *(const v8h*)(row + 16 + kh * 8);
            b[j] = __builtin_shufflevector(lo, hi, 0,1,2,3,4,5,6,7,8,9,10,11,12,13,14,15);
        }
        #pragma unroll
        for (int i = 0; i < 2; ++i)
            #pragma unroll
            for (int j = 0; j < 2; ++j)
                acc[i][j] = __builtin_amdgcn_wmma_f32_16x16x32_f16(
                    false, a[i], false, b[j], (short)0, acc[i][j], false, false);
    };

    stage(0, 0);
    __syncthreads();
    int cur = 0;
    for (int s = 0; s < NSTEPS; ++s) {
        if (s + 1 < NSTEPS) stage(cur ^ 1, (s + 1) * BK);
        compute(cur);
        __syncthreads();
        cur ^= 1;
    }

    // Epilogue: D layout is VGPR r -> M = r + 8*kh, lanes -> N = col.
    #pragma unroll
    for (int i = 0; i < 2; ++i) {
        #pragma unroll
        for (int j = 0; j < 2; ++j) {
            const int hcol = n0 + wn * 32 + j * 16 + col;
            #pragma unroll
            for (int r = 0; r < 8; ++r) {
                const int row = m0 + wm * 32 + i * 16 + r + kh * 8;
                accOut[(size_t)row * H_DIM + hcol] = (_Float16)acc[i][j][r];
            }
        }
    }
}

// ---------------------------------------------------------------------------
// Tail: per position, stm-dependent concat order, clamp^2 activation, dot with
// the selected bucket's W_out row only, wave32 reduction.
// ---------------------------------------------------------------------------
__global__ __launch_bounds__(256) void nnue_tail(
    const _Float16* __restrict__ accW, const _Float16* __restrict__ accB,
    const int* __restrict__ is_white,  const int* __restrict__ buckets,
    const float* __restrict__ Wout,    const float* __restrict__ bout,
    float* __restrict__ out)
{
    const int wave = threadIdx.x >> 5;
    const int lane = threadIdx.x & 31;
    const int p = blockIdx.x * 8 + wave;
    if (p >= B_POS) return;

    const bool white  = is_white[p] != 0;
    const int  bucket = buckets[p];
    const _Float16* first  = (white ? accW : accB) + (size_t)p * H_DIM;
    const _Float16* second = (white ? accB : accW) + (size_t)p * H_DIM;
    const float* wo = Wout + (size_t)bucket * (2 * H_DIM);

    float sum = 0.f;
    #pragma unroll
    for (int tstep = 0; tstep < 4; ++tstep) {
        const int j = tstep * 256 + lane * 8;
        v8h h1 = *(const v8h*)(first + j);
        v8h h2 = *(const v8h*)(second + j);
        float4 w10 = *(const float4*)(wo + j);
        float4 w11 = *(const float4*)(wo + j + 4);
        float4 w20 = *(const float4*)(wo + H_DIM + j);
        float4 w21 = *(const float4*)(wo + H_DIM + j + 4);
        const float wv1[8] = {w10.x, w10.y, w10.z, w10.w, w11.x, w11.y, w11.z, w11.w};
        const float wv2[8] = {w20.x, w20.y, w20.z, w20.w, w21.x, w21.y, w21.z, w21.w};
        #pragma unroll
        for (int e = 0; e < 8; ++e) {
            float x1 = (float)h1[e]; x1 = x1 < 0.f ? 0.f : (x1 > 1.f ? 1.f : x1);
            float x2 = (float)h2[e]; x2 = x2 < 0.f ? 0.f : (x2 > 1.f ? 1.f : x2);
            sum += x1 * x1 * wv1[e] + x2 * x2 * wv2[e];
        }
    }
    #pragma unroll
    for (int off = 16; off > 0; off >>= 1)
        sum += __shfl_xor(sum, off, 32);
    if (lane == 0) out[p] = sum + bout[bucket];
}

// ---------------------------------------------------------------------------
extern "C" void kernel_launch(void* const* d_in, const int* in_sizes, int n_in,
                              void* d_out, int out_size, void* d_ws, size_t ws_size,
                              hipStream_t stream)
{
    const float* fw       = (const float*)d_in[0];
    const float* fb       = (const float*)d_in[1];
    const int*   is_white = (const int*)  d_in[2];
    const int*   buckets  = (const int*)  d_in[3];
    const float* Ww       = (const float*)d_in[4];
    const float* bw       = (const float*)d_in[5];
    const float* Wb       = (const float*)d_in[6];
    const float* bb       = (const float*)d_in[7];
    const float* Wout     = (const float*)d_in[8];
    const float* bout     = (const float*)d_in[9];

    _Float16* accW = (_Float16*)d_ws;
    _Float16* accB = accW + (size_t)B_POS * H_DIM;   // 2 * 32 MB f16 scratch

    const int gemm_blocks = (B_POS / BM) * 2 * (H_DIM / BN);  // 256 * 2 * 8 = 4096
    ft_gemm_f16<<<gemm_blocks, 256, 0, stream>>>(fw, fb, Ww, Wb, bw, bb, accW, accB);
    nnue_tail<<<B_POS / 8, 256, 0, stream>>>(accW, accB, is_white, buckets, Wout, bout,
                                             (float*)d_out);
}
